// TransformerBlockMHC_4260607557641
// MI455X (gfx1250) — matrix operand order repaired and checked by reading, not compiled, not run on hardware
//
#include <hip/hip_runtime.h>
#include <hip/hip_bf16.h>
#include <math.h>

typedef _Float16 f16;
typedef __attribute__((ext_vector_type(16))) _Float16 v16h;
typedef __attribute__((ext_vector_type(8)))  _Float16 v8h;
typedef __attribute__((ext_vector_type(8)))  float    v8f;

#define EMBED 768
#define HEADS 12
#define HEAD_DIM 64
#define SEQ 1024
#define BATCH 8
#define TOKENS (BATCH * SEQ)
#define STREAMS 4
#define CHUNK (EMBED / STREAMS)   // 192
#define MLP_HIDDEN 3072
#define QK3 (3 * EMBED)           // 2304

// ---------------- workspace layout (bytes) ----------------
static constexpr size_t OFF_SK    = 0;                                   // 48 f32
static constexpr size_t OFF_WQKV  = 256;
static constexpr size_t SZ_WQKV   = (size_t)QK3 * EMBED * 2;
static constexpr size_t OFF_WPROJ = OFF_WQKV + SZ_WQKV;
static constexpr size_t SZ_WPROJ  = (size_t)EMBED * EMBED * 2;
static constexpr size_t OFF_WFC1  = OFF_WPROJ + SZ_WPROJ;
static constexpr size_t SZ_WFC1   = (size_t)MLP_HIDDEN * EMBED * 2;
static constexpr size_t OFF_WFC2  = OFF_WFC1 + SZ_WFC1;
static constexpr size_t SZ_WFC2   = SZ_WFC1;
static constexpr size_t OFF_HA    = OFF_WFC2 + SZ_WFC2;                  // f16 activations (reused)
static constexpr size_t SZ_HA     = (size_t)TOKENS * EMBED * 2;
static constexpr size_t OFF_BIG   = OFF_HA + SZ_HA;                      // qkv f32, later hmid f16
static constexpr size_t SZ_BIG    = (size_t)TOKENS * QK3 * 4;
static constexpr size_t OFF_Q     = OFF_BIG + SZ_BIG;
static constexpr size_t SZ_QKV16  = (size_t)TOKENS * EMBED * 2;
static constexpr size_t OFF_K     = OFF_Q + SZ_QKV16;
static constexpr size_t OFF_VT    = OFF_K + SZ_QKV16;
static constexpr size_t OFF_CTX   = OFF_VT + SZ_QKV16;
static constexpr size_t OFF_T     = OFF_CTX + SZ_QKV16;                  // f32 transform output
static constexpr size_t SZ_T      = (size_t)TOKENS * EMBED * 4;
static constexpr size_t OFF_X1    = OFF_T + SZ_T;                        // f32 mid residual

// ---------------- helpers ----------------
__device__ inline v16h cat16(v8h lo, v8h hi) {
  return __builtin_shufflevector(lo, hi, 0,1,2,3,4,5,6,7,8,9,10,11,12,13,14,15);
}
// A fragment (16x32 f16): lane m = lane&15, khalf = lane>>4.
// halves at k = khalf*8 + [0..7] and k = 16 + khalf*8 + [0..7]  (p already at row + k0 + khalf*8)
__device__ inline v16h load_a_frag(const f16* p) {
  v8h lo = *(const v8h*)p;
  v8h hi = *(const v8h*)(p + 16);
  return cat16(lo, hi);
}
// B fragment (32x16 f16): lane n = lane&15; halves at k = khalf*8 + [0..7] and k = 16 + khalf*8 + [0..7] (p already at k0 + khalf*8)
__device__ inline v16h load_b_frag(const f16* p) {
  v8h lo = *(const v8h*)p;
  v8h hi = *(const v8h*)(p + 16);
  return cat16(lo, hi);
}

// ---------------- tiny kernels ----------------
__global__ __launch_bounds__(32) void k_sinkhorn(const float* __restrict__ hres,
                                                 const float* __restrict__ hpre,
                                                 const float* __restrict__ hpost,
                                                 float* __restrict__ out /*24 f32*/) {
  if (threadIdx.x != 0) return;
  float A[4][4];
  for (int r = 0; r < 4; ++r)
    for (int c = 0; c < 4; ++c) A[r][c] = __expf(hres[r * 4 + c]);
  for (int it = 0; it < 20; ++it) {
    for (int r = 0; r < 4; ++r) {
      float s = A[r][0] + A[r][1] + A[r][2] + A[r][3] + 1e-8f;
      for (int c = 0; c < 4; ++c) A[r][c] /= s;
    }
    for (int c = 0; c < 4; ++c) {
      float s = A[0][c] + A[1][c] + A[2][c] + A[3][c] + 1e-8f;
      for (int r = 0; r < 4; ++r) A[r][c] /= s;
    }
  }
  for (int r = 0; r < 4; ++r)
    for (int c = 0; c < 4; ++c) out[r * 4 + c] = A[r][c];
  {
    float m = fmaxf(fmaxf(hpre[0], hpre[1]), fmaxf(hpre[2], hpre[3]));
    float e[4], s = 0.f;
    for (int i = 0; i < 4; ++i) { e[i] = __expf(hpre[i] - m); s += e[i]; }
    for (int i = 0; i < 4; ++i) out[16 + i] = e[i] / s;
  }
  {
    float m = fmaxf(fmaxf(hpost[0], hpost[1]), fmaxf(hpost[2], hpost[3]));
    float e[4], s = 0.f;
    for (int i = 0; i < 4; ++i) { e[i] = __expf(hpost[i] - m); s += e[i]; }
    for (int i = 0; i < 4; ++i) out[20 + i] = e[i] / s;
  }
}

__global__ __launch_bounds__(256) void k_f32_to_f16(const float* __restrict__ src,
                                                    f16* __restrict__ dst, int n) {
  int i = blockIdx.x * 256 + threadIdx.x;
  if (i < n) dst[i] = (f16)src[i];
}

// per-token: x_pre = sum_n Hpre[n]*x[..n,c], tile, layernorm, -> f16
__global__ __launch_bounds__(256) void k_pre_ln(const float* __restrict__ xin,
                                                const float* __restrict__ sk_pre,
                                                const float* __restrict__ g,
                                                const float* __restrict__ bta,
                                                f16* __restrict__ hout) {
  int t = blockIdx.x, tid = threadIdx.x;
  __shared__ float p[CHUNK];
  __shared__ float red[256], red2[256];
  float w0 = sk_pre[0], w1 = sk_pre[1], w2 = sk_pre[2], w3 = sk_pre[3];
  float val = 0.f;
  if (tid < CHUNK) {
    const float* xr = xin + (size_t)t * EMBED;
    val = w0 * xr[tid] + w1 * xr[CHUNK + tid] + w2 * xr[2 * CHUNK + tid] + w3 * xr[3 * CHUNK + tid];
    p[tid] = val;
  }
  red[tid]  = (tid < CHUNK) ? val : 0.f;
  red2[tid] = (tid < CHUNK) ? val * val : 0.f;
  __syncthreads();
  for (int s = 128; s > 0; s >>= 1) {
    if (tid < s) { red[tid] += red[tid + s]; red2[tid] += red2[tid + s]; }
    __syncthreads();
  }
  float mu = red[0] * (1.f / CHUNK);
  float var = red2[0] * (1.f / CHUNK) - mu * mu;
  float inv = rsqrtf(var + 1e-5f);
  for (int d = tid; d < EMBED; d += 256) {
    float y = (p[d % CHUNK] - mu) * inv * g[d] + bta[d];
    hout[(size_t)t * EMBED + d] = (f16)y;
  }
}

// ---------------- WMMA GEMM: out[M,N] = A[M,K](f16) @ W[N,K]^T + bias ----------------
// EPI 0: f32 store with bias. EPI 1: bias + exact GELU, f16 store.
template <int EPI>
__global__ __launch_bounds__(32) void k_gemm(const f16* __restrict__ A,
                                             const f16* __restrict__ Bw,
                                             const float* __restrict__ bias,
                                             void* __restrict__ outv,
                                             int M, int N, int K) {
  const int lane = threadIdx.x;
  const int mrow = lane & 15, khalf = lane >> 4;
  const int n0 = blockIdx.x * 64;
  const int m0 = blockIdx.y * 32;
  v8f acc[2][4];
#pragma unroll
  for (int mt = 0; mt < 2; ++mt)
#pragma unroll
    for (int nt = 0; nt < 4; ++nt) acc[mt][nt] = (v8f)0.f;

  for (int k0 = 0; k0 < K; k0 += 32) {
    v16h a[2], b[4];
#pragma unroll
    for (int mt = 0; mt < 2; ++mt)
      a[mt] = load_a_frag(A + (size_t)(m0 + mt * 16 + mrow) * K + k0 + khalf * 8);
#pragma unroll
    for (int nt = 0; nt < 4; ++nt)
      b[nt] = load_b_frag(Bw + (size_t)(n0 + nt * 16 + mrow) * K + k0 + khalf * 8);
#pragma unroll
    for (int mt = 0; mt < 2; ++mt)
#pragma unroll
      for (int nt = 0; nt < 4; ++nt)
        acc[mt][nt] = __builtin_amdgcn_wmma_f32_16x16x32_f16(
            false, a[mt], false, b[nt], (short)0, acc[mt][nt], false, false);
  }
#pragma unroll
  for (int mt = 0; mt < 2; ++mt)
#pragma unroll
    for (int nt = 0; nt < 4; ++nt)
#pragma unroll
      for (int v = 0; v < 8; ++v) {
        int row = m0 + mt * 16 + khalf * 8 + v;
        int col = n0 + nt * 16 + mrow;
        float xv = acc[mt][nt][v] + bias[col];
        if (EPI == 0) {
          ((float*)outv)[(size_t)row * N + col] = xv;
        } else {
          float gx = 0.5f * xv * (1.0f + erff(xv * 0.70710678118654752f));
          ((f16*)outv)[(size_t)row * N + col] = (f16)gx;
        }
      }
}

// ---------------- split qkv(f32) -> Q*scale, K (b,h,n,d) f16 ; V transposed (b,h,d,n) f16 ----
__global__ __launch_bounds__(256) void k_split(const float* __restrict__ qkv,
                                               f16* __restrict__ Q, f16* __restrict__ Kk,
                                               f16* __restrict__ Vt) {
  size_t i = (size_t)blockIdx.x * 256 + threadIdx.x;
  if (i >= (size_t)TOKENS * EMBED) return;
  int col = (int)(i % EMBED);
  size_t tok = i / EMBED;
  int h = col >> 6, d = col & 63;
  int b = (int)(tok >> 10), nn = (int)(tok & 1023);
  size_t base = tok * QK3;
  float qv = qkv[base + col];
  float kv = qkv[base + EMBED + col];
  float vv = qkv[base + 2 * EMBED + col];
  size_t bh = (size_t)(b * HEADS + h);
  Q[(bh * SEQ + nn) * HEAD_DIM + d]  = (f16)(qv * 0.125f);   // 64^-0.5
  Kk[(bh * SEQ + nn) * HEAD_DIM + d] = (f16)kv;
  Vt[(bh * HEAD_DIM + d) * SEQ + nn] = (f16)vv;
}

// ---------------- flash attention: one wave handles 32 q rows of one (b,h) ----------------
// Softmax strategy: lane r owns q-row (q0 + r). S tiles are dumped C-layout -> LDS f32;
// each lane streams its own contiguous row back, does max/exp/sum lane-locally (no
// cross-lane ops), writes P as packed f16; only alpha / rs scalars are bpermute-broadcast.
__global__ __launch_bounds__(32) void k_attn(const f16* __restrict__ Q,
                                             const f16* __restrict__ Kk,
                                             const f16* __restrict__ Vt,
                                             f16* __restrict__ ctx) {
  __shared__ float slds[32 * 32];   // raw scores, row-major [qrow][key]
  __shared__ f16   plds[32 * 32];   // exp'd probs,  row-major [qrow][key]
  const int lane = threadIdx.x;
  const int mrow = lane & 15, khalf = lane >> 4;
  const int bh = blockIdx.y, b = bh / HEADS, h = bh % HEADS;
  const int q0 = blockIdx.x * 32;
  const f16* Qb = Q  + (size_t)bh * SEQ * HEAD_DIM;
  const f16* Kb = Kk + (size_t)bh * SEQ * HEAD_DIM;
  const f16* Vb = Vt + (size_t)bh * HEAD_DIM * SEQ;

  v16h aQ[2][2];
#pragma unroll
  for (int qt = 0; qt < 2; ++qt)
#pragma unroll
    for (int ks = 0; ks < 2; ++ks)
      aQ[qt][ks] = load_a_frag(Qb + (size_t)(q0 + qt * 16 + mrow) * HEAD_DIM + ks * 32 + khalf * 8);

  v8f O[2][4];
#pragma unroll
  for (int qt = 0; qt < 2; ++qt)
#pragma unroll
    for (int nt = 0; nt < 4; ++nt) O[qt][nt] = (v8f)0.f;
  float rm_r = -1e30f;   // running max of the q-row this lane owns
  float rs_r = 0.f;      // running sum of the q-row this lane owns

  for (int kb = 0; kb < SEQ; kb += 32) {
    v16h bK[2][2];
#pragma unroll
    for (int kt = 0; kt < 2; ++kt)
#pragma unroll
      for (int ks = 0; ks < 2; ++ks)
        bK[kt][ks] = load_b_frag(Kb + (size_t)(kb + kt * 16 + mrow) * HEAD_DIM + ks * 32 + khalf * 8);

    v8f S[2][2];
#pragma unroll
    for (int qt = 0; qt < 2; ++qt)
#pragma unroll
      for (int kt = 0; kt < 2; ++kt) {
        v8f z = (v8f)0.f;
        z = __builtin_amdgcn_wmma_f32_16x16x32_f16(false, aQ[qt][0], false, bK[kt][0], (short)0, z, false, false);
        z = __builtin_amdgcn_wmma_f32_16x16x32_f16(false, aQ[qt][1], false, bK[kt][1], (short)0, z, false, false);
        S[qt][kt] = z;
      }

    // C-layout -> LDS (f32 raw scores)
#pragma unroll
    for (int qt = 0; qt < 2; ++qt)
#pragma unroll
      for (int kt = 0; kt < 2; ++kt)
#pragma unroll
        for (int v = 0; v < 8; ++v)
          slds[(qt * 16 + khalf * 8 + v) * 32 + kt * 16 + mrow] = S[qt][kt][v];
    asm volatile("s_wait_dscnt 0x0" ::: "memory");

    // lane-local softmax over own row
    float pv[32];
    {
      const float4* rowp = (const float4*)&slds[lane * 32];
#pragma unroll
      for (int c = 0; c < 8; ++c) {
        float4 q4 = rowp[c];
        pv[c * 4 + 0] = q4.x; pv[c * 4 + 1] = q4.y;
        pv[c * 4 + 2] = q4.z; pv[c * 4 + 3] = q4.w;
      }
    }
    float mx = rm_r;
#pragma unroll
    for (int i = 0; i < 32; ++i) mx = fmaxf(mx, pv[i]);
    float alpha_r = __expf(rm_r - mx);
    rm_r = mx;
    float psum = 0.f;
#pragma unroll
    for (int i = 0; i < 32; ++i) { pv[i] = __expf(pv[i] - mx); psum += pv[i]; }
    rs_r = rs_r * alpha_r + psum;
    // write P back packed f16
    {
      f16* pw = &plds[lane * 32];
#pragma unroll
      for (int c = 0; c < 4; ++c) {
        v8h u;
#pragma unroll
        for (int j = 0; j < 8; ++j) u[j] = (f16)pv[c * 8 + j];
        *(v8h*)(pw + c * 8) = u;
      }
    }
    asm volatile("s_wait_dscnt 0x0" ::: "memory");

    // broadcast alpha to C-layout lanes, rescale accumulators
#pragma unroll
    for (int qt = 0; qt < 2; ++qt)
#pragma unroll
      for (int v = 0; v < 8; ++v) {
        float av = __shfl(alpha_r, qt * 16 + khalf * 8 + v, 32);
#pragma unroll
        for (int nt = 0; nt < 4; ++nt) O[qt][nt][v] *= av;
      }

    v16h aP[2], bV[4];
#pragma unroll
    for (int qt = 0; qt < 2; ++qt)
      aP[qt] = load_a_frag(&plds[(qt * 16 + mrow) * 32 + khalf * 8]);
#pragma unroll
    for (int nt = 0; nt < 4; ++nt)
      bV[nt] = load_b_frag(Vb + (size_t)(nt * 16 + mrow) * SEQ + kb + khalf * 8);
#pragma unroll
    for (int qt = 0; qt < 2; ++qt)
#pragma unroll
      for (int nt = 0; nt < 4; ++nt)
        O[qt][nt] = __builtin_amdgcn_wmma_f32_16x16x32_f16(
            false, aP[qt], false, bV[nt], (short)0, O[qt][nt], false, false);
  }

  // epilogue: broadcast row sums, normalize, store f16 context (token, H*hd)
#pragma unroll
  for (int qt = 0; qt < 2; ++qt)
#pragma unroll
    for (int v = 0; v < 8; ++v) {
      float rsv = __shfl(rs_r, qt * 16 + khalf * 8 + v, 32);
      float inv = 1.0f / rsv;
      int row = q0 + qt * 16 + khalf * 8 + v;
      int col = h * HEAD_DIM + mrow;
#pragma unroll
      for (int nt = 0; nt < 4; ++nt) {
        float o = O[qt][nt][v] * inv;
        ctx[((size_t)(b * SEQ + row)) * EMBED + col + nt * 16] = (f16)o;
      }
    }
}

// ---------------- mhc combine: out = Hres@xs + Hpost * mean_n(t) ----------------
__global__ __launch_bounds__(256) void k_combine(const float* __restrict__ xin,
                                                 const float* __restrict__ t,
                                                 const float* __restrict__ hres,
                                                 const float* __restrict__ hpost,
                                                 float* __restrict__ out) {
  size_t i = (size_t)blockIdx.x * 256 + threadIdx.x;
  if (i >= (size_t)TOKENS * CHUNK) return;
  int c = (int)(i % CHUNK);
  size_t tok = i / CHUNK;
  const float* xr = xin + tok * EMBED;
  const float* tr = t + tok * EMBED;
  float xs0 = xr[c], xs1 = xr[CHUNK + c], xs2 = xr[2 * CHUNK + c], xs3 = xr[3 * CHUNK + c];
  float tm = 0.25f * (tr[c] + tr[CHUNK + c] + tr[2 * CHUNK + c] + tr[3 * CHUNK + c]);
  float* orow = out + tok * EMBED;
#pragma unroll
  for (int n = 0; n < 4; ++n) {
    float acc = hres[n * 4 + 0] * xs0 + hres[n * 4 + 1] * xs1 +
                hres[n * 4 + 2] * xs2 + hres[n * 4 + 3] * xs3 + hpost[n] * tm;
    orow[n * CHUNK + c] = acc;
  }
}

// ---------------- launch ----------------
extern "C" void kernel_launch(void* const* d_in, const int* in_sizes, int n_in,
                              void* d_out, int out_size, void* d_ws, size_t ws_size,
                              hipStream_t stream) {
  const float* x      = (const float*)d_in[0];
  const float* ln1_g  = (const float*)d_in[1];
  const float* ln1_b  = (const float*)d_in[2];
  const float* ln2_g  = (const float*)d_in[3];
  const float* ln2_b  = (const float*)d_in[4];
  const float* qkv_w  = (const float*)d_in[5];
  const float* qkv_b  = (const float*)d_in[6];
  const float* proj_w = (const float*)d_in[7];
  const float* proj_b = (const float*)d_in[8];
  const float* fc1_w  = (const float*)d_in[9];
  const float* fc1_b  = (const float*)d_in[10];
  const float* fc2_w  = (const float*)d_in[11];
  const float* fc2_b  = (const float*)d_in[12];
  const float* a_hres = (const float*)d_in[13];
  const float* a_hpre = (const float*)d_in[14];
  const float* a_hpost= (const float*)d_in[15];
  const float* m_hres = (const float*)d_in[16];
  const float* m_hpre = (const float*)d_in[17];
  const float* m_hpost= (const float*)d_in[18];

  char* ws = (char*)d_ws;
  float* sk    = (float*)(ws + OFF_SK);
  f16* wqkv    = (f16*)(ws + OFF_WQKV);
  f16* wproj   = (f16*)(ws + OFF_WPROJ);
  f16* wfc1    = (f16*)(ws + OFF_WFC1);
  f16* wfc2    = (f16*)(ws + OFF_WFC2);
  f16* hA      = (f16*)(ws + OFF_HA);
  float* qkv32 = (float*)(ws + OFF_BIG);
  f16* hmid    = (f16*)(ws + OFF_BIG);   // reuse big arena after attention
  f16* Qh      = (f16*)(ws + OFF_Q);
  f16* Kh      = (f16*)(ws + OFF_K);
  f16* Vth     = (f16*)(ws + OFF_VT);
  f16* ctx     = (f16*)(ws + OFF_CTX);
  float* tbuf  = (float*)(ws + OFF_T);
  float* x1    = (float*)(ws + OFF_X1);
  float* outp  = (float*)d_out;

  // weights -> f16
  k_f32_to_f16<<<(QK3 * EMBED + 255) / 256, 256, 0, stream>>>(qkv_w, wqkv, QK3 * EMBED);
  k_f32_to_f16<<<(EMBED * EMBED + 255) / 256, 256, 0, stream>>>(proj_w, wproj, EMBED * EMBED);
  k_f32_to_f16<<<(MLP_HIDDEN * EMBED + 255) / 256, 256, 0, stream>>>(fc1_w, wfc1, MLP_HIDDEN * EMBED);
  k_f32_to_f16<<<(EMBED * MLP_HIDDEN + 255) / 256, 256, 0, stream>>>(fc2_w, wfc2, EMBED * MLP_HIDDEN);

  // sinkhorn + softmaxes (attention set at sk[0..23], mlp set at sk[24..47])
  k_sinkhorn<<<1, 32, 0, stream>>>(a_hres, a_hpre, a_hpost, sk);
  k_sinkhorn<<<1, 32, 0, stream>>>(m_hres, m_hpre, m_hpost, sk + 24);

  // ===== attention mhc =====
  k_pre_ln<<<TOKENS, 256, 0, stream>>>(x, sk + 16, ln1_g, ln1_b, hA);
  {
    dim3 g(QK3 / 64, TOKENS / 32);
    k_gemm<0><<<g, 32, 0, stream>>>(hA, wqkv, qkv_b, qkv32, TOKENS, QK3, EMBED);
  }
  k_split<<<((size_t)TOKENS * EMBED + 255) / 256, 256, 0, stream>>>(qkv32, Qh, Kh, Vth);
  {
    dim3 g(SEQ / 32, BATCH * HEADS);
    k_attn<<<g, 32, 0, stream>>>(Qh, Kh, Vth, ctx);
  }
  {
    dim3 g(EMBED / 64, TOKENS / 32);
    k_gemm<0><<<g, 32, 0, stream>>>(ctx, wproj, proj_b, tbuf, TOKENS, EMBED, EMBED);
  }
  k_combine<<<((size_t)TOKENS * CHUNK + 255) / 256, 256, 0, stream>>>(x, tbuf, sk, sk + 20, x1);

  // ===== mlp mhc =====
  k_pre_ln<<<TOKENS, 256, 0, stream>>>(x1, sk + 24 + 16, ln2_g, ln2_b, hA);
  {
    dim3 g(MLP_HIDDEN / 64, TOKENS / 32);
    k_gemm<1><<<g, 32, 0, stream>>>(hA, wfc1, fc1_b, hmid, TOKENS, MLP_HIDDEN, EMBED);
  }
  {
    dim3 g(EMBED / 64, TOKENS / 32);
    k_gemm<0><<<g, 32, 0, stream>>>(hmid, wfc2, fc2_b, tbuf, TOKENS, EMBED, MLP_HIDDEN);
  }
  k_combine<<<((size_t)TOKENS * CHUNK + 255) / 256, 256, 0, stream>>>(x1, tbuf, sk + 24, sk + 24 + 20, outp);

  (void)in_sizes; (void)n_in; (void)out_size; (void)ws_size;
}
